// Capsule_41644002902348
// MI455X (gfx1250) — compile-verified, matrix operand order checked by reading, not run on hardware
//
#include <hip/hip_runtime.h>

// ---------------------------------------------------------------------------
// Capsule routing for MI455X (gfx1250, wave32, WMMA).
//
//   u_vecs : (128, 1024, 256) f32      W : (256, 512) f32
//   u_hat  = u_vecs @ W  -> stored f16 as [b][n][i][d]  (134 MB)
//   3 routing iterations (softmax over n=16, reduce over i=1024, squash)
//
// GEMM runs on v_wmma_f32_16x16x32_f16 (f32 accumulate). Routing iteration 0
// (b=0 -> uniform c) is fused into the GEMM epilogue. Iterations 1..2 are one
// fused pass each over u_hat. All reductions deterministic (no atomics).
// ---------------------------------------------------------------------------

typedef __attribute__((ext_vector_type(16))) _Float16 v16h;
typedef __attribute__((ext_vector_type(8)))  _Float16 v8h;
typedef __attribute__((ext_vector_type(8)))  float    v8f;

#define BATCH 128
#define IC    1024      // input capsules
#define KDIM  256       // input dim
#define MDIM  512       // NC*DC
#define NC    16
#define DC    32
#define NSLC  8         // reduction slices (i-tiles / i-splits)

// ---------------------------------------------------------------------------
// GEMM: per block computes a 128(i) x 128(m) tile of u_hat for one batch.
// K chunked by 64 through LDS (fp32 -> f16 convert on the fly).
// 8 waves: wave = (wrow 0..3, wcol 0..1), each owns 2x4 grid of 16x16 tiles.
// Epilogue: store f16 u_hat [b][n][i][d] + deterministic per-i-tile column
// sums (= iteration-0 partial outputs, since c is uniform when b==0).
// ---------------------------------------------------------------------------
__global__ __launch_bounds__(256)
void caps_gemm(const float* __restrict__ u, const float* __restrict__ w,
               _Float16* __restrict__ uh, float* __restrict__ pslices)
{
    __shared__ _Float16 As[128][72];   // 64 k + 8 pad (16B-aligned rows)
    __shared__ _Float16 Bs[128][72];   // W tile transposed: [m][k]
    __shared__ float    csum[128][16]; // deterministic column reduction

    const int tid = threadIdx.x;
    const int bid = blockIdx.x;            // 128 b * 8 it * 4 mt = 4096
    const int b   = bid >> 5;
    const int it  = (bid >> 2) & 7;
    const int mt  = bid & 3;
    const int i0  = it * 128;
    const int m0  = mt * 128;

    const int lane = tid & 31;
    const int wave = tid >> 5;
    const int wrow = wave >> 1;            // 0..3 -> 32-row band
    const int wcol = wave & 1;             // 0..1 -> 64-col band
    const int r0   = lane & 15;
    const int half = lane >> 4;            // 0/1
    const int koff = half * 8;

    v8f acc[2][4];
#pragma unroll
    for (int tm = 0; tm < 2; ++tm)
#pragma unroll
        for (int tn = 0; tn < 4; ++tn)
#pragma unroll
            for (int q = 0; q < 8; ++q) acc[tm][tn][q] = 0.0f;

    for (int kc = 0; kc < KDIM; kc += 64) {
        __syncthreads();
        // ---- load A chunk: 128 rows x 64 k (float2 -> 2x f16) ----
#pragma unroll
        for (int r = 0; r < 16; ++r) {
            const int f   = tid + 256 * r;        // 4096 float2 total
            const int row = f >> 5;
            const int kp  = (f & 31) * 2;
            const float2 v = *reinterpret_cast<const float2*>(
                u + ((size_t)b * IC + i0 + row) * KDIM + kc + kp);
            As[row][kp]     = (_Float16)v.x;
            As[row][kp + 1] = (_Float16)v.y;
        }
        // ---- load B chunk transposed: Bs[m][k] ----
#pragma unroll
        for (int r = 0; r < 32; ++r) {
            const int e  = tid + 256 * r;         // 8192 elements
            const int k  = e >> 7;
            const int mc = e & 127;
            Bs[mc][k] = (_Float16)w[(size_t)(kc + k) * MDIM + m0 + mc];
        }
        __syncthreads();

#pragma unroll
        for (int ks = 0; ks < 2; ++ks) {          // two K=32 WMMA steps
            const int kb = ks * 32;
            v16h afr[2], bfr[4];
#pragma unroll
            for (int tm = 0; tm < 2; ++tm) {
                const int row = wrow * 32 + tm * 16 + r0;
#pragma unroll
                for (int t = 0; t < 8; ++t) {
                    afr[tm][t]     = As[row][kb + koff + t];
                    afr[tm][8 + t] = As[row][kb + 16 + koff + t];
                }
            }
#pragma unroll
            for (int tn = 0; tn < 4; ++tn) {
                const int col = wcol * 64 + tn * 16 + r0;
#pragma unroll
                for (int t = 0; t < 8; ++t) {
                    bfr[tn][t]     = Bs[col][kb + koff + t];
                    bfr[tn][8 + t] = Bs[col][kb + 16 + koff + t];
                }
            }
#pragma unroll
            for (int tm = 0; tm < 2; ++tm)
#pragma unroll
                for (int tn = 0; tn < 4; ++tn)
                    acc[tm][tn] = __builtin_amdgcn_wmma_f32_16x16x32_f16(
                        false, afr[tm], false, bfr[tn],
                        (short)0, acc[tm][tn], false, false);
        }
    }
    __syncthreads();

    // ---- epilogue: u_hat f16 + per-column partial sums over this i-tile ----
#pragma unroll
    for (int tm = 0; tm < 2; ++tm) {
#pragma unroll
        for (int tn = 0; tn < 4; ++tn) {
            const int col = wcol * 64 + tn * 16 + r0;
            const int m   = m0 + col;
            const int n   = m >> 5;
            const int d   = m & 31;
            float ps = 0.0f;
#pragma unroll
            for (int v = 0; v < 8; ++v) {
                // C layout: VGPR v holds M = v + half*8, N = r0
                const int i = i0 + wrow * 32 + tm * 16 + half * 8 + v;
                uh[(((size_t)b * NC + n) * IC + i) * DC + d] =
                    (_Float16)acc[tm][tn][v];
                ps += acc[tm][tn][v];
            }
            csum[col][wrow * 4 + tm * 2 + half] = ps;  // unique slot
        }
    }
    __syncthreads();
    if (tid < 128) {
        float s = 0.0f;
#pragma unroll
        for (int q = 0; q < 16; ++q) s += csum[tid][q];
        pslices[(size_t)it * (BATCH * MDIM) + (size_t)b * MDIM + m0 + tid] = s;
    }
}

// ---------------------------------------------------------------------------
// Fused routing pass: for each i -> b[n]=dot(out_prev, u_hat), softmax over n,
// accumulate c[n]*u_hat. One block = (batch, split of 128 i's). Lane layout:
// n = lane>>1, d-half = (lane&1)*16; wave-level shfl reductions give the dot
// (xor 1) and the 16-capsule softmax (xor 2,4,8,16). Per-wave accumulators
// cover the full 16x32 output; deterministic cross-wave LDS reduction.
// ---------------------------------------------------------------------------
__global__ __launch_bounds__(256)
void caps_route(const _Float16* __restrict__ uh,
                const float* __restrict__ outPrev,
                float* __restrict__ pslices)
{
    __shared__ float red[512][8];

    const int tid   = threadIdx.x;
    const int b     = blockIdx.x >> 3;
    const int split = blockIdx.x & 7;
    const int lane  = tid & 31;
    const int wave  = tid >> 5;
    const int n     = lane >> 1;
    const int dbase = (lane & 1) * 16;

    float ov[16];
    const float* op = outPrev + ((size_t)b * NC + n) * DC + dbase;
#pragma unroll
    for (int t = 0; t < 16; ++t) ov[t] = op[t];

    float acc[16];
#pragma unroll
    for (int t = 0; t < 16; ++t) acc[t] = 0.0f;

    const _Float16* ubase = uh + ((size_t)b * NC + n) * IC * DC + dbase;
    const int ibase = split * 128 + wave * 16;

    for (int k = 0; k < 16; ++k) {
        const _Float16* p = ubase + (size_t)(ibase + k) * DC;
        const v8h u0 = *reinterpret_cast<const v8h*>(p);
        const v8h u1 = *reinterpret_cast<const v8h*>(p + 8);
        float uf[16];
#pragma unroll
        for (int t = 0; t < 8; ++t) { uf[t] = (float)u0[t]; uf[8 + t] = (float)u1[t]; }

        float pd = 0.0f;
#pragma unroll
        for (int t = 0; t < 16; ++t) pd += ov[t] * uf[t];
        pd += __shfl_xor(pd, 1, 32);              // full 32-d dot -> b[n]

        float mx = pd;                            // softmax over 16 capsules
        mx = fmaxf(mx, __shfl_xor(mx, 2, 32));
        mx = fmaxf(mx, __shfl_xor(mx, 4, 32));
        mx = fmaxf(mx, __shfl_xor(mx, 8, 32));
        mx = fmaxf(mx, __shfl_xor(mx, 16, 32));
        const float e = __expf(pd - mx);
        float s = e;
        s += __shfl_xor(s, 2, 32);
        s += __shfl_xor(s, 4, 32);
        s += __shfl_xor(s, 8, 32);
        s += __shfl_xor(s, 16, 32);
        const float c = e / s;
#pragma unroll
        for (int t = 0; t < 16; ++t) acc[t] += c * uf[t];
    }

#pragma unroll
    for (int t = 0; t < 16; ++t) red[n * DC + dbase + t][wave] = acc[t];
    __syncthreads();

    float s0 = 0.0f, s1 = 0.0f;
#pragma unroll
    for (int wv = 0; wv < 8; ++wv) { s0 += red[tid][wv]; s1 += red[tid + 256][wv]; }
    float* dst = pslices + (size_t)split * (BATCH * MDIM) + (size_t)b * MDIM;
    dst[tid]       = s0;
    dst[tid + 256] = s1;
}

// ---------------------------------------------------------------------------
// Sum the 8 deterministic slices, scale, squash: x / sqrt(sum_d x^2 + 1e-7).
// 32 consecutive elements = one (b,n) capsule -> wave shfl reduction.
// ---------------------------------------------------------------------------
__global__ __launch_bounds__(256)
void caps_squash(const float* __restrict__ pslices, float* __restrict__ dst,
                 float scale)
{
    const int idx = blockIdx.x * 256 + threadIdx.x;   // 0 .. 65535
    float x = 0.0f;
#pragma unroll
    for (int s = 0; s < NSLC; ++s) x += pslices[(size_t)s * (BATCH * MDIM) + idx];
    x *= scale;
    float sq = x * x;
    sq += __shfl_xor(sq, 1, 32);
    sq += __shfl_xor(sq, 2, 32);
    sq += __shfl_xor(sq, 4, 32);
    sq += __shfl_xor(sq, 8, 32);
    sq += __shfl_xor(sq, 16, 32);
    dst[idx] = x / sqrtf(sq + 1e-7f);
}

// ---------------------------------------------------------------------------
extern "C" void kernel_launch(void* const* d_in, const int* in_sizes, int n_in,
                              void* d_out, int out_size, void* d_ws, size_t ws_size,
                              hipStream_t stream)
{
    const float* u = (const float*)d_in[0];   // (128,1024,256) f32
    const float* w = (const float*)d_in[1];   // (256,512) f32
    float* out = (float*)d_out;               // (128,16,32) f32

    char* ws = (char*)d_ws;
    const size_t uhBytes = (size_t)BATCH * NC * IC * DC * sizeof(_Float16); // 134 MB
    const size_t slBytes = (size_t)NSLC * BATCH * MDIM * sizeof(float);     // 2 MB
    _Float16* uh      = (_Float16*)ws;
    float*    pslices = (float*)(ws + uhBytes);
    float*    outPrev = (float*)(ws + uhBytes + slBytes);                   // 256 KB

    // GEMM + fused routing iteration 0 partials (uniform c = 1/16)
    caps_gemm<<<4096, 256, 0, stream>>>(u, w, uh, pslices);
    caps_squash<<<256, 256, 0, stream>>>(pslices, outPrev, 1.0f / 16.0f);
    // routing iteration 1
    caps_route<<<1024, 256, 0, stream>>>(uh, outPrev, pslices);
    caps_squash<<<256, 256, 0, stream>>>(pslices, outPrev, 1.0f);
    // routing iteration 2 (final)
    caps_route<<<1024, 256, 0, stream>>>(uh, outPrev, pslices);
    caps_squash<<<256, 256, 0, stream>>>(pslices, out, 1.0f);
}
